// VAD_Model_81192061763619
// MI455X (gfx1250) — compile-verified, matrix operand order checked by reading, not run on hardware
//
#include <hip/hip_runtime.h>
#include <hip/hip_bf16.h>
#include <math.h>

// ---------------------------------------------------------------------------
// Model dims (from reference): B=32, T=512, H=2048, DS=32, DC=4, MEM=128,
// NC=1, PD=128.  MT = B*T = 16384.
// ---------------------------------------------------------------------------
#define BB   32
#define TT   512
#define HH   2048
#define DSS  32
#define DCC  4
#define MEMN 128
#define PDN  128
#define MT   (BB * TT)

// Global (device memory) address space — forces global_load/global_store
// instead of flat_* (flat ties up DScnt and the LDS aperture checks).
#define AS_G __attribute__((address_space(1)))

typedef __attribute__((ext_vector_type(16))) __bf16        v16bf;
typedef __attribute__((ext_vector_type(8)))  float         v8f;
typedef __attribute__((ext_vector_type(4)))  unsigned int  u32x4;  // POD 128-bit

__device__ __forceinline__ unsigned short f2bf(float f) {
    unsigned int u = __float_as_uint(f);
    u += 0x7FFFu + ((u >> 16) & 1u);   // round-to-nearest-even
    return (unsigned short)(u >> 16);
}
__device__ __forceinline__ float bf2f(unsigned short h) {
    return __uint_as_float(((unsigned int)h) << 16);
}
__device__ __forceinline__ float silu(float x) {
    return x / (1.f + __expf(-x));
}

// ---------------------------------------------------------------------------
// f32 -> bf16 conversion (grid-stride)
// ---------------------------------------------------------------------------
__global__ void cvt_bf16_kernel(const float* __restrict__ src,
                                unsigned short* __restrict__ dst, size_t n) {
    const AS_G float* gs = (const AS_G float*)src;
    AS_G unsigned short* gd = (AS_G unsigned short*)dst;
    size_t i = (size_t)blockIdx.x * blockDim.x + threadIdx.x;
    size_t st = (size_t)gridDim.x * blockDim.x;
    for (; i < n; i += st) gd[i] = f2bf(gs[i]);
}

// ---------------------------------------------------------------------------
// bf16 WMMA GEMM:  C[M,N] = A[M,K] @ B[N,K]^T + bias[N]   (K = HH = 2048)
// A, B row-major bf16 (ld = K).  Workgroup = 256 threads = 8 waves,
// WG tile 128x128, wave tile 32x64 (2x4 WMMA 16x16x32 tiles).
// Row-major [row,K] fragment layout is identical for A and B operands:
//   lane: row = base + lane%16; K-half kh = (lane/16)*8
//   frag elems 0..7  = src[row, k+kh    .. +7]   (one global_load_b128)
//   frag elems 8..15 = src[row, k+16+kh .. +7]   (one global_load_b128)
// ---------------------------------------------------------------------------
__device__ __forceinline__ v16bf ld_frag(const unsigned short* p) {
    const AS_G u32x4* gp = (const AS_G u32x4*)p;
    union { v16bf v; u32x4 q[2]; } u;
    u.q[0] = gp[0];   // K = k+kh    .. +7  (16B)
    u.q[1] = gp[2];   // K = k+16+kh .. +7  (16B, +32B offset)
    return u.v;
}

template <bool OUT_BF16>
__global__ __launch_bounds__(256)
void gemm_bf16_wmma(const unsigned short* __restrict__ A,
                    const unsigned short* __restrict__ B,
                    const float* __restrict__ bias,
                    void* __restrict__ Cout,
                    int M, int N, int ldc) {
    constexpr int K = HH;
    const int lane  = threadIdx.x & 31;
    const int wave  = threadIdx.x >> 5;
    const int wm    = wave & 3;          // 0..3 (M)
    const int wn    = wave >> 2;         // 0..1 (N)
    const int mBase = blockIdx.x * 128 + wm * 32;
    const int nBase = blockIdx.y * 128 + wn * 64;
    const int l16   = lane & 15;
    const int kh    = (lane >> 4) << 3;

    v8f acc[2][4];
#pragma unroll
    for (int i = 0; i < 2; ++i)
#pragma unroll
        for (int j = 0; j < 4; ++j)
#pragma unroll
            for (int e = 0; e < 8; ++e) acc[i][j][e] = 0.f;

    // Clamped base pointers (keeps EXEC all-ones for WMMA; stores are guarded)
    const unsigned short* pa[2];
    const unsigned short* pb[4];
#pragma unroll
    for (int i = 0; i < 2; ++i) {
        int r = mBase + i * 16 + l16; if (r > M - 1) r = M - 1;
        pa[i] = A + (size_t)r * K + kh;
    }
#pragma unroll
    for (int j = 0; j < 4; ++j) {
        int c = nBase + j * 16 + l16; if (c > N - 1) c = N - 1;
        pb[j] = B + (size_t)c * K + kh;
    }

#pragma unroll 2
    for (int k = 0; k < K; k += 32) {
        v16bf a[2], b[4];
#pragma unroll
        for (int i = 0; i < 2; ++i) a[i] = ld_frag(pa[i]);
#pragma unroll
        for (int j = 0; j < 4; ++j) b[j] = ld_frag(pb[j]);
        // prefetch next K-slab (generic ptr form: device-legal, lowers to prefetch_b8)
        __builtin_prefetch(pa[0] + 256, 0, 1);
        __builtin_prefetch(pb[0] + 256, 0, 1);
#pragma unroll
        for (int i = 0; i < 2; ++i)
#pragma unroll
            for (int j = 0; j < 4; ++j)
                acc[i][j] = __builtin_amdgcn_wmma_f32_16x16x32_bf16(
                    false, a[i], false, b[j], (short)0, acc[i][j], false, false);
#pragma unroll
        for (int i = 0; i < 2; ++i) pa[i] += 32;
#pragma unroll
        for (int j = 0; j < 4; ++j) pb[j] += 32;
    }

    // Store D: lane holds col = l16, rows r0..r0+7 (VGPR e), per ISA C/D layout
    const AS_G float* gbias = (const AS_G float*)bias;
#pragma unroll
    for (int i = 0; i < 2; ++i) {
#pragma unroll
        for (int j = 0; j < 4; ++j) {
            int col = nBase + j * 16 + l16;
            if (col >= N) continue;
            float bv = bias ? gbias[col] : 0.f;
            int r0 = mBase + i * 16 + ((lane >> 4) << 3);
#pragma unroll
            for (int e = 0; e < 8; ++e) {
                int r = r0 + e;
                if (r < M) {
                    float v = acc[i][j][e] + bv;
                    if (OUT_BF16)
                        ((AS_G unsigned short*)Cout)[(size_t)r * ldc + col] = f2bf(v);
                    else
                        ((AS_G float*)Cout)[(size_t)r * ldc + col] = v;
                }
            }
        }
    }
}

// ---------------------------------------------------------------------------
// Depthwise causal conv (K=4) + SiLU.  xin/xout: (B*T, H) bf16.
// out[b,t,h] = silu( sum_k w[h,k]*x[b,t-3+k,h] + cb[h] )
// ---------------------------------------------------------------------------
__global__ void conv_silu_kernel(const unsigned short* __restrict__ xin,
                                 const float* __restrict__ cw,
                                 const float* __restrict__ cb,
                                 unsigned short* __restrict__ xout,
                                 long long total) {
    const AS_G unsigned short* gx = (const AS_G unsigned short*)xin;
    const AS_G float* gw = (const AS_G float*)cw;
    const AS_G float* gb = (const AS_G float*)cb;
    AS_G unsigned short* go = (AS_G unsigned short*)xout;
    long long i  = (long long)blockIdx.x * blockDim.x + threadIdx.x;
    long long st = (long long)gridDim.x * blockDim.x;
    for (; i < total; i += st) {
        int h = (int)(i % HH);
        long long bt = i / HH;
        int t = (int)(bt % TT);
        float s = gb[h];
#pragma unroll
        for (int k = 0; k < DCC; ++k) {
            int tt = t - (DCC - 1) + k;
            if (tt >= 0)
                s += bf2f(gx[i + (long long)(k - (DCC - 1)) * HH]) * gw[h * DCC + k];
        }
        go[i] = f2bf(silu(s));
    }
}

// ---------------------------------------------------------------------------
// SSM scan, exact (general A).  One wave per batch; lane d owns state h[d].
//   h_t = h_{t-1} @ A^T + xs_t * Bp   ;   cvals[b] = h_{T-1} . Cp
// ---------------------------------------------------------------------------
__global__ __launch_bounds__(32)
void ssm_scan_kernel(const float* __restrict__ xs,   // (B*T, DS)
                     const float* __restrict__ A,    // (DS, DS)
                     const float* __restrict__ Bp,
                     const float* __restrict__ Cp,
                     float* __restrict__ cvals) {
    int b = blockIdx.x;
    int d = threadIdx.x;                 // 0..31
    float Arow[DSS];
#pragma unroll
    for (int j = 0; j < DSS; ++j) Arow[j] = A[d * DSS + j];   // hnew[d]=Σ h[j]A[d,j]
    float bpd = Bp[d], cpd = Cp[d];
    float h = 0.f;
    const float* xsb = xs + (size_t)b * TT * DSS;
    for (int t = 0; t < TT; ++t) {
        float hn = 0.f;
#pragma unroll
        for (int j = 0; j < DSS; ++j) hn += Arow[j] * __shfl(h, j, 32);
        h = hn + xsb[t * DSS + d] * bpd;
    }
    float c = h * cpd;
#pragma unroll
    for (int m = 16; m >= 1; m >>= 1) c += __shfl_xor(c, m, 32);
    if (d == 0) cvals[b] = c;
}

// ---------------------------------------------------------------------------
// Gate at t=T-1 only (f32 for accuracy): silu( x[b,T-1,:] @ W_in[H+h,:] + b )
// ---------------------------------------------------------------------------
__global__ __launch_bounds__(256)
void gate_last_kernel(const float* __restrict__ x,
                      const float* __restrict__ W_in,
                      const float* __restrict__ b_in,
                      float* __restrict__ gout) {
    __shared__ float xr[HH];
    int b = blockIdx.y;
    const float* xrow = x + ((size_t)b * TT + (TT - 1)) * HH;
    for (int i = threadIdx.x; i < HH; i += blockDim.x) xr[i] = xrow[i];
    __syncthreads();
    int h = blockIdx.x * blockDim.x + threadIdx.x;   // 0..H-1
    const float* w = W_in + (size_t)(HH + h) * HH;
    float s = b_in[HH + h];
    for (int k = 0; k < HH; ++k) s += xr[k] * w[k];
    gout[(size_t)b * HH + h] = silu(s);
}

// ---------------------------------------------------------------------------
// ylast[b,h] = (cvals[b] + Dp[h]) * xh_conv[b,T-1,h] * gate_silu[b,h]  (bf16)
// ---------------------------------------------------------------------------
__global__ void final_pre_kernel(const unsigned short* __restrict__ xh_conv,
                                 const float* __restrict__ gate_silu,
                                 const float* __restrict__ cvals,
                                 const float* __restrict__ Dp,
                                 unsigned short* __restrict__ ylast) {
    int i = blockIdx.x * blockDim.x + threadIdx.x;   // < B*H
    int b = i / HH, h = i % HH;
    float xv = bf2f(xh_conv[((size_t)b * TT + (TT - 1)) * HH + h]);
    ylast[i] = f2bf((cvals[b] + Dp[h]) * xv * gate_silu[i]);
}

// ---------------------------------------------------------------------------
// Head: score (min L2 distance to pm/sm rows), fc out, proj.  One block / batch.
// d_out layout: out[0..31] | score[32..63] | proj[64 .. 64+32*128)
// ---------------------------------------------------------------------------
__global__ __launch_bounds__(256)
void head_kernel(const float* __restrict__ feat,     // (B, H)
                 const float* __restrict__ pm,
                 const float* __restrict__ sm,
                 const float* __restrict__ W_fc, const float* __restrict__ b_fc,
                 const float* __restrict__ W_proj, const float* __restrict__ b_proj,
                 float* __restrict__ out) {
    __shared__ float fr[HH];
    __shared__ float red[256];
    int b = blockIdx.x, tid = threadIdx.x;
    for (int i = tid; i < HH; i += 256) fr[i] = feat[(size_t)b * HH + i];
    __syncthreads();

    // 128 pm rows + 128 sm rows -> one row per thread
    const float* row = (tid < MEMN) ? (pm + (size_t)tid * HH)
                                    : (sm + (size_t)(tid - MEMN) * HH);
    float d2 = 0.f;
    for (int k = 0; k < HH; ++k) { float d = row[k] - fr[k]; d2 += d * d; }
    red[tid] = d2;
    __syncthreads();
    for (int s = 128; s >= 1; s >>= 1) {
        if (tid < s) red[tid] = fminf(red[tid], red[tid + s]);
        __syncthreads();
    }
    if (tid == 0) out[BB + b] = sqrtf(red[0]);

    if (tid < PDN) {
        const float* w = W_proj + (size_t)tid * HH;
        float s = b_proj[tid];
        for (int k = 0; k < HH; ++k) s += fr[k] * w[k];
        out[2 * BB + (size_t)b * PDN + tid] = s;
    } else if (tid == PDN) {
        float s = b_fc[0];
        for (int k = 0; k < HH; ++k) s += fr[k] * W_fc[k];
        out[b] = s;
    }
}

// ---------------------------------------------------------------------------
// Launch
// ---------------------------------------------------------------------------
extern "C" void kernel_launch(void* const* d_in, const int* in_sizes, int n_in,
                              void* d_out, int out_size, void* d_ws, size_t ws_size,
                              hipStream_t stream) {
    const float* x      = (const float*)d_in[0];
    const float* W_in   = (const float*)d_in[1];
    const float* b_in   = (const float*)d_in[2];
    const float* conv_w = (const float*)d_in[3];
    const float* conv_b = (const float*)d_in[4];
    const float* W_ssm  = (const float*)d_in[5];
    const float* b_ssm  = (const float*)d_in[6];
    const float* Amat   = (const float*)d_in[7];
    const float* Bp     = (const float*)d_in[8];
    const float* Cp     = (const float*)d_in[9];
    const float* Dp     = (const float*)d_in[10];
    const float* W_out  = (const float*)d_in[11];
    const float* b_out  = (const float*)d_in[12];
    const float* pm     = (const float*)d_in[13];
    const float* sm     = (const float*)d_in[14];
    const float* W_fc   = (const float*)d_in[15];
    const float* b_fc   = (const float*)d_in[16];
    const float* W_proj = (const float*)d_in[17];
    const float* b_proj = (const float*)d_in[18];
    float* out = (float*)d_out;

    // Workspace layout (256B aligned)
    char* base = (char*)d_ws;
    size_t off = 0;
    auto take = [&](size_t bytes) { size_t o = off; off += (bytes + 255) & ~(size_t)255; return o; };
    unsigned short* x_bf      = (unsigned short*)(base + take((size_t)MT * HH * 2));     // 64MB
    unsigned short* win_bf    = (unsigned short*)(base + take((size_t)HH * HH * 2));     // 8MB (xh half)
    unsigned short* wssm_bf   = (unsigned short*)(base + take((size_t)DSS * HH * 2));
    unsigned short* wout_bf   = (unsigned short*)(base + take((size_t)HH * HH * 2));     // 8MB
    unsigned short* xh_raw    = (unsigned short*)(base + take((size_t)MT * HH * 2));     // 64MB
    unsigned short* xh_conv   = (unsigned short*)(base + take((size_t)MT * HH * 2));     // 64MB
    float*          xs        = (float*)        (base + take((size_t)MT * DSS * 4));     // 2MB
    float*          cvals     = (float*)        (base + take((size_t)BB * 4));
    float*          gate_silu = (float*)        (base + take((size_t)BB * HH * 4));
    unsigned short* ylast     = (unsigned short*)(base + take((size_t)BB * HH * 2));
    float*          feat      = (float*)        (base + take((size_t)BB * HH * 4));

    // 1) f32 -> bf16 conversions
    cvt_bf16_kernel<<<4096, 256, 0, stream>>>(x, x_bf, (size_t)MT * HH);
    cvt_bf16_kernel<<<2048, 256, 0, stream>>>(W_in, win_bf, (size_t)HH * HH);     // rows 0..H-1
    cvt_bf16_kernel<<<256, 256, 0, stream>>>(W_ssm, wssm_bf, (size_t)DSS * HH);
    cvt_bf16_kernel<<<2048, 256, 0, stream>>>(W_out, wout_bf, (size_t)HH * HH);

    // 2) GEMM1: xh_raw = x @ W_in[0:H].T + b_in[0:H]   (M=16384, N=2048, K=2048)
    gemm_bf16_wmma<true><<<dim3(MT / 128, HH / 128), 256, 0, stream>>>(
        x_bf, win_bf, b_in, (void*)xh_raw, MT, HH, HH);

    // 3) depthwise causal conv + SiLU
    conv_silu_kernel<<<8192, 256, 0, stream>>>(xh_raw, conv_w, conv_b, xh_conv,
                                               (long long)MT * HH);

    // 4) xs = xh_conv @ W_ssm.T + b_ssm   (M=16384, N=32, K=2048)
    gemm_bf16_wmma<false><<<dim3(MT / 128, 1), 256, 0, stream>>>(
        xh_conv, wssm_bf, b_ssm, (void*)xs, MT, DSS, DSS);

    // 5) exact SSM scan -> cvals[b] = h_{T-1} . Cp
    ssm_scan_kernel<<<BB, 32, 0, stream>>>(xs, Amat, Bp, Cp, cvals);

    // 6) gate at t=T-1 (f32)
    gate_last_kernel<<<dim3(HH / 256, BB), 256, 0, stream>>>(x, W_in, b_in, gate_silu);

    // 7) ylast = (c_b + Dp) * xh_conv[:,T-1,:] * silu(gate)
    final_pre_kernel<<<(BB * HH) / 256, 256, 0, stream>>>(xh_conv, gate_silu, cvals,
                                                          Dp, ylast);

    // 8) feat = ylast @ W_out.T + b_out   (M=32, N=2048, K=2048)
    gemm_bf16_wmma<false><<<dim3(1, HH / 128), 256, 0, stream>>>(
        ylast, wout_bf, b_out, (void*)feat, BB, HH, HH);

    // 9) head: fc out, min-distance score, proj
    head_kernel<<<BB, 256, 0, stream>>>(feat, pm, sm, W_fc, b_fc, W_proj, b_proj, out);

    (void)in_sizes; (void)n_in; (void)out_size; (void)ws_size;
}
